// _MultiResInteractionLayer_75118978007443
// MI455X (gfx1250) — compile-verified, matrix operand order checked by reading, not run on hardware
//
#include <hip/hip_runtime.h>
#include <hip/hip_bf16.h>
#include <math.h>

// ---------------- types ----------------
typedef __bf16 bf16_t;
typedef __attribute__((ext_vector_type(8)))  bf16_t v8bf;
typedef __attribute__((ext_vector_type(16))) bf16_t v16bf;
typedef __attribute__((ext_vector_type(8)))  float  v8f;

#define S_DIM    128
#define V_DIM    32
#define WIDTH    320          // S + 6V
#define GDIM     480          // 2S + 7V (valid gate columns)
#define GPAD     512          // padded N for the g GEMM (multiple of 64)
#define PDIM     1120         // packed operand width for edge-message block-GEMM
#define N_ATOM   16000
#define N_FRAG   2000
#define N_PROT   8000

// GEMM epilogue mode bits
#define MB_BIAS 1
#define MB_SILU 2
#define MB_F32  4
#define MB_SCAT 8

// ---------------- WMMA GEMM: D = A(bf16, MxK) * Bt(bf16, NxK)^T ----------------
// One 16x64 output tile per wave32 (4 accumulators): the A fragment is loaded
// once per K-chunk and reused by 4 WMMAs, quadrupling FLOPs per A byte.
__launch_bounds__(256)
__global__ void gemm_bf16_wmma4(const bf16_t* __restrict__ A, int lda,
                                const bf16_t* __restrict__ Bt, int ldb,
                                int Mtiles, int Ngroups, int K,
                                const float* __restrict__ bias,
                                float* __restrict__ outF, bf16_t* __restrict__ outB, int ldo,
                                const int* __restrict__ dstIdx, float* __restrict__ agg,
                                int Mvalid, int mode)
{
  int wave = blockIdx.x * (blockDim.x >> 5) + (threadIdx.x >> 5);
  if (wave >= Mtiles * Ngroups) return;          // wave-uniform: EXEC stays all-ones for WMMA
  int tm = wave / Ngroups;
  int tg = wave % Ngroups;
  int lane  = threadIdx.x & 31;
  int rrow  = lane & 15;
  int khalf = (lane >> 4) << 3;                  // 0 or 8 (A/B bf16 fragment layout)

  const bf16_t* ap = A  + (size_t)(tm * 16 + rrow) * lda + khalf;
  const bf16_t* bp = Bt + (size_t)(tg * 64 + rrow) * ldb + khalf;
  const size_t bstep = (size_t)16 * ldb;

  v8f acc[4] = {};
  for (int k0 = 0; k0 < K; k0 += 32) {
    v8bf a0 = *(const v8bf*)(ap + k0);
    v8bf a1 = *(const v8bf*)(ap + k0 + 16);
    __builtin_prefetch(ap + k0 + 256, 0, 3);     // global_prefetch_b8 (streamed A operand)
    v16bf av = __builtin_shufflevector(a0, a1, 0,1,2,3,4,5,6,7,8,9,10,11,12,13,14,15);
#pragma unroll
    for (int j = 0; j < 4; ++j) {
      const bf16_t* bj = bp + (size_t)j * bstep + k0;
      v8bf b0 = *(const v8bf*)(bj);
      v8bf b1 = *(const v8bf*)(bj + 16);
      v16bf bv = __builtin_shufflevector(b0, b1, 0,1,2,3,4,5,6,7,8,9,10,11,12,13,14,15);
      acc[j] = __builtin_amdgcn_wmma_f32_16x16x32_bf16(false, av, false, bv, (short)0, acc[j], false, false);
    }
  }

  int mofs = (lane >> 4) << 3;                   // C/D layout: VGPR i, lanes>=16 hold M=8+i
#pragma unroll
  for (int j = 0; j < 4; ++j) {
    int n = tg * 64 + j * 16 + (lane & 15);
#pragma unroll
    for (int i = 0; i < 8; ++i) {
      int m = tm * 16 + i + mofs;
      float v = acc[j][i];
      if (mode & MB_BIAS) v += bias[n];
      if (mode & MB_SILU) v = v / (1.f + __expf(-v));
      if (mode & MB_SCAT) {
        if (m < Mvalid) atomicAdd(&agg[(size_t)dstIdx[m] * WIDTH + n], v);
      } else if (mode & MB_F32) {
        outF[(size_t)m * ldo + n] = v;
      } else {
        outB[(size_t)m * ldo + n] = (bf16_t)v;
      }
    }
  }
}

// ---------------- edge-scalar builder (gather + RBF), bf16 output ----------------
__global__ void build_es_kernel(bf16_t* __restrict__ ES, int Kpad, int E,
    const float* __restrict__ posP, const int* __restrict__ ip,
    const float* __restrict__ posM, const int* __restrict__ im,
    const float* __restrict__ triRef, float* __restrict__ nvecOut,
    const float* f0, int s0, const int* i0, const int* n0,
    const float* f1, int s1, const int* i1, const int* n1,
    const float* f2, int s2, const int* i2, const int* n2,
    const float* f3, int s3, const int* i3, const int* n3,
    int nb)
{
  int e = blockIdx.x, tid = threadIdx.x;
  bf16_t* row = ES + (size_t)e * Kpad;
  if (e >= E) {
    for (int c = tid; c < Kpad; c += 128) row[c] = (bf16_t)0.f;
    if (tid < 3) nvecOut[e * 3 + tid] = 0.f;
    return;
  }
  int pi = ip ? ip[e] : e;
  int mi = im ? im[e] : e;
  float dx = posP[pi * 3 + 0] - posM[mi * 3 + 0];
  float dy = posP[pi * 3 + 1] - posM[mi * 3 + 1];
  float dz = posP[pi * 3 + 2] - posM[mi * 3 + 2];
  float d  = sqrtf(dx * dx + dy * dy + dz * dz);
  float inv = 1.f / (d + 1e-8f);
  if (tid == 0) nvecOut[e * 3 + 0] = dx * inv;
  if (tid == 1) nvecOut[e * 3 + 1] = dy * inv;
  if (tid == 2) nvecOut[e * 3 + 2] = dz * inv;
  float dd = triRef ? (d - triRef[e]) : 0.f;
  int base = triRef ? 33 : 16;
  int featEnd = base + nb * 128;
  const float gam = 2.56f;                        // (N_RBF/CUTOFF)^2
  for (int c = tid; c < Kpad; c += 128) {
    float val = 0.f;
    if (c < 16) { float t = d - (10.f / 15.f) * c; val = __expf(-gam * t * t); }
    else if (triRef && c < 32) { float t = fabsf(dd) - (10.f / 15.f) * (c - 16); val = __expf(-gam * t * t); }
    else if (triRef && c == 32) { val = dd; }
    else if (c >= base && c < featEnd) {
      int b = (c - base) >> 7, j = (c - base) & 127;
      const float* f; int st; const int* ix; const int* nd;
      if      (b == 0) { f = f0; st = s0; ix = i0; nd = n0; }
      else if (b == 1) { f = f1; st = s1; ix = i1; nd = n1; }
      else if (b == 2) { f = f2; st = s2; ix = i2; nd = n2; }
      else             { f = f3; st = s3; ix = i3; nd = n3; }
      int r = ix ? ix[e] : e;
      if (nd) r = nd[r];
      val = f[(size_t)r * st + j];
    }
    row[c] = (bf16_t)val;
  }
}

// ---------------- packed operand builder for edge-message block GEMM ----------------
// P columns: [0,128) ss*gA | [128,160) dot1*gB | [160,544) 3x (ss*gC)*nvec_c |
// [544,1120) 3x { vs1*gD, cross2*gE, m1*gH, vs2*gF, cross1*gG, m2*gI } (32 each)
__global__ void build_P_kernel(bf16_t* __restrict__ P, int E,
    const float* __restrict__ hsrc, const int* __restrict__ srcIdx,
    const bf16_t* __restrict__ G, const float* __restrict__ nvec)
{
  __shared__ float ss[128];
  __shared__ float vv[192];                       // v1 comps [0,96), v2 comps [96,192)
  __shared__ float nv[3], dot1s[32], dot2s[32];
  int e = blockIdx.x, tid = threadIdx.x;
  bf16_t* row = P + (size_t)e * PDIM;
  if (e >= E) { for (int c = tid; c < PDIM; c += 128) row[c] = (bf16_t)0.f; return; }
  int r = srcIdx ? srcIdx[e] : e;
  const float* h = hsrc + (size_t)r * WIDTH;
  ss[tid] = h[tid];
  if (tid < 96) { vv[tid] = h[128 + tid]; vv[96 + tid] = h[224 + tid]; }
  if (tid < 3) nv[tid] = nvec[e * 3 + tid];
  __syncthreads();
  if (tid < 32) {
    dot1s[tid] = vv[tid*3]*nv[0] + vv[tid*3+1]*nv[1] + vv[tid*3+2]*nv[2];
    dot2s[tid] = vv[96+tid*3]*nv[0] + vv[96+tid*3+1]*nv[1] + vv[96+tid*3+2]*nv[2];
  }
  __syncthreads();
  const bf16_t* g = G + (size_t)e * GPAD;
  for (int c = tid; c < PDIM; c += 128) {
    float val;
    if (c < 128) val = ss[c] * (float)g[c];
    else if (c < 160) { int v = c - 128; val = dot1s[v] * (float)g[128 + v]; }
    else if (c < 544) { int q = c - 160; int cc = q >> 7, j = q & 127;
                        val = ss[j] * (float)g[160 + j] * nv[cc]; }
    else {
      int q = c - 544; int cc = q / 192; int rr = q % 192; int grp = rr >> 5; int v = rr & 31;
      const float* a1 = &vv[v * 3]; const float* a2 = &vv[96 + v * 3];
      switch (grp) {
        case 0: val = a1[cc] * (float)g[288 + v]; break;
        case 1: { float cx = (cc==0) ? a2[1]*nv[2]-a2[2]*nv[1]
                          : (cc==1) ? a2[2]*nv[0]-a2[0]*nv[2]
                                    : a2[0]*nv[1]-a2[1]*nv[0];
                  val = cx * (float)g[320 + v]; } break;
        case 2: val = (nv[cc]*dot1s[v] - a1[cc]*(1.f/3.f)) * (float)g[352 + v]; break;
        case 3: val = a2[cc] * (float)g[384 + v]; break;
        case 4: { float cx = (cc==0) ? a1[1]*nv[2]-a1[2]*nv[1]
                          : (cc==1) ? a1[2]*nv[0]-a1[0]*nv[2]
                                    : a1[0]*nv[1]-a1[1]*nv[0];
                  val = cx * (float)g[416 + v]; } break;
        default: val = (nv[cc]*dot2s[v] - a2[cc]*(1.f/3.f)) * (float)g[448 + v]; break;
      }
    }
    row[c] = (bf16_t)val;
  }
}

// ---------------- weight preprocessing ----------------
// out[n][k] (bf16, Npad x Kpad) = in[k][n] (f32, K x Nsrc), zero-padded in both dims
__global__ void transpose_pad(bf16_t* __restrict__ out, const float* __restrict__ in,
                              int K, int Nsrc, int Npad, int Kpad)
{
  int t = blockIdx.x * blockDim.x + threadIdx.x;
  if (t >= Npad * Kpad) return;
  int n = t / Kpad, k = t % Kpad;
  out[t] = (bf16_t)((k < K && n < Nsrc) ? in[(size_t)k * Nsrc + n] : 0.f);
}

// Block-structured 1120x320 edge-message weight, stored transposed (320 x 1120)
__global__ void build_bigwt(bf16_t* __restrict__ out,
  const float* Wss, const float* Wvs, const float* Wsv,
  const float* Wv1, const float* We1, const float* Wh1,
  const float* Wv2, const float* We2, const float* Wh2)
{
  int t = blockIdx.x * blockDim.x + threadIdx.x;
  if (t >= 320 * PDIM) return;
  int n = t / PDIM, k = t % PDIM;
  float val = 0.f;
  if (n < 128) {
    if (k < 128) val = Wss[k * 128 + n];
    else if (k < 160) val = Wvs[(k - 128) * 128 + n];
  } else {
    int isv2 = (n >= 224);
    int nn = n - (isv2 ? 224 : 128);
    int v = nn / 3, c = nn % 3;
    if (!isv2 && k >= 160 && k < 544) {
      int cc = (k - 160) >> 7, j = (k - 160) & 127;
      if (cc == c) val = Wsv[j * 32 + v];
    } else if (k >= 544) {
      int q = k - 544; int cc = q / 192, rr = q % 192, grp = rr >> 5, j = rr & 31;
      if (cc == c) {
        if (!isv2) { if (grp == 0) val = Wv1[j*32+v]; else if (grp == 1) val = We1[j*32+v]; else if (grp == 2) val = Wh1[j*32+v]; }
        else       { if (grp == 3) val = Wv2[j*32+v]; else if (grp == 4) val = We2[j*32+v]; else if (grp == 5) val = Wh2[j*32+v]; }
      }
    }
  }
  out[t] = (bf16_t)val;
}

// Block-structured 320x320 irrep_linear weight, stored transposed
__global__ void build_bigpt(bf16_t* __restrict__ out,
  const float* Ps, const float* Pv1, const float* Pv2)
{
  int t = blockIdx.x * blockDim.x + threadIdx.x;
  if (t >= 320 * 320) return;
  int n = t / 320, k = t % 320;
  float val = 0.f;
  if (n < 128) { if (k < 128) val = Ps[k * 128 + n]; }
  else {
    int isv2 = (n >= 224);
    int nn = n - (isv2 ? 224 : 128);
    int w = nn / 3, c = nn % 3;
    int kb = isv2 ? 224 : 128;
    if (k >= kb && k < kb + 96) {
      int kk = k - kb; int v = kk / 3, cc = kk % 3;
      if (cc == c) val = (isv2 ? Pv2 : Pv1)[v * 32 + w];
    }
  }
  out[t] = (bf16_t)val;
}

// ---------------- small elementwise helpers ----------------
__global__ void cast_f32_bf16(bf16_t* __restrict__ out, const float* __restrict__ in, int n)
{
  int t = blockIdx.x * blockDim.x + threadIdx.x;
  if (t < n) out[t] = (bf16_t)in[t];
}

__global__ void build_cond_atom(bf16_t* __restrict__ out, const float* __restrict__ tea,
                                const float* __restrict__ hfrag, const int* __restrict__ fragid, int n)
{
  int t = blockIdx.x * blockDim.x + threadIdx.x;
  if (t >= n * 256) return;
  int i = t >> 8, c = t & 255;
  float v = (c < 128) ? tea[(size_t)i * 128 + c]
                      : hfrag[(size_t)fragid[i] * WIDTH + (c - 128)];
  out[t] = (bf16_t)v;
}

// ---------------- residual + scalar_act + adaLN ----------------
__device__ __forceinline__ float blk_reduce(float v, float* sh, int tid)
{
  sh[tid] = v; __syncthreads();
  for (int o = 64; o > 0; o >>= 1) { if (tid < o) sh[tid] += sh[tid + o]; __syncthreads(); }
  float r = sh[0]; __syncthreads();
  return r;
}

__global__ void node_update_kernel(const float* __restrict__ hOld,
                                   const float* __restrict__ proj,
                                   const float* __restrict__ mod,
                                   float* __restrict__ out)
{
  __shared__ float sh[128];
  int i = blockIdx.x, tid = threadIdx.x;
  const float* hr = hOld + (size_t)i * WIDTH;
  const float* pr = proj + (size_t)i * WIDTH;
  const float* mr = mod  + (size_t)i * WIDTH;
  float* orow = out + (size_t)i * WIDTH;

  float p = pr[tid];
  float s = hr[tid] + p / (1.f + __expf(-p));                   // residual + silu(scalar)
  float v0 = hr[128 + tid] + pr[128 + tid];                     // vectors: plain residual
  float v1 = (tid < 64) ? (hr[256 + tid] + pr[256 + tid]) : 0.f;

  float mu  = blk_reduce(s, sh, tid) * (1.f / 128.f);
  float d0  = s - mu;
  float var = blk_reduce(d0 * d0, sh, tid) * (1.f / 128.f);
  float rstd = rsqrtf(var + 1e-6f);
  float vsum = blk_reduce(v0 * v0 + v1 * v1, sh, tid);
  float vr = sqrtf(vsum * (1.f / 64.f) + 1e-6f);

  orow[tid] = d0 * rstd * (1.f + mr[tid]) + mr[128 + tid];
  orow[128 + tid] = v0 / vr * (1.f + mr[256 + tid / 3]);
  if (tid < 64) orow[256 + tid] = v1 / vr * (1.f + mr[256 + (128 + tid) / 3]);
}

// ---------------- host orchestration ----------------
extern "C" void kernel_launch(void* const* d_in, const int* in_sizes, int n_in,
                              void* d_out, int out_size, void* d_ws, size_t ws_size,
                              hipStream_t stream)
{
  (void)in_sizes; (void)n_in; (void)out_size; (void)ws_size;

  enum { I_HATOM=0,I_HFRAG,I_HPROT,I_APOS,I_TFRAG,I_PPOS,I_FRAGID,
         I_RASRC,I_RADST,I_BSRC,I_BDST,I_CUTS,I_CUTD,I_PFSRC,I_PFDST,
         I_FFSRC,I_FFDST,I_TRISRC,I_TRIDST,I_TRIREF,I_TEA,I_TEF,
         I_RAP=22, I_AAP=34, I_TRIP=46, I_AFP=58, I_PFP=70, I_FFP=82,
         I_APROJ=94, I_FPROJ=97, I_ALN=100, I_FLN=102 };
  // conv param leaf order: w1,b1,w2,Wss,Wvs,Wsv,Wv1,We1,Wh1,Wv2,We2,Wh2

  auto F  = [&](int i) { return (const float*)d_in[i]; };
  auto II = [&](int i) { return (const int*)d_in[i]; };

  const float* hatom = F(I_HATOM); const float* hfrag = F(I_HFRAG); const float* hprot = F(I_HPROT);
  const float* apos = F(I_APOS);  const float* tfr = F(I_TFRAG);   const float* ppos = F(I_PPOS);
  const int* fragid = II(I_FRAGID);
  const float* tea = F(I_TEA);    const float* tef = F(I_TEF);

  float* outAtom = (float*)d_out;
  float* outFrag = (float*)d_out + (size_t)N_ATOM * WIDTH;

  // workspace carve (256B aligned)
  char* ws = (char*)d_ws;
  size_t off = 0;
  auto carve = [&](size_t bytes) -> char* {
    off = (off + 255) & ~(size_t)255;
    char* p = ws + off; off += bytes; return p;
  };
  bf16_t* ES    = (bf16_t*)carve((size_t)40000 * 576 * 2);
  bf16_t* HID   = (bf16_t*)carve((size_t)40000 * 128 * 2);
  bf16_t* G     = (bf16_t*)carve((size_t)40000 * GPAD * 2);
  char*   Pbase = carve((size_t)40000 * PDIM * 2);
  bf16_t* P     = (bf16_t*)Pbase;
  float*  NVEC  = (float*)carve((size_t)40000 * 3 * 4);
  float*  AGG_A = (float*)carve((size_t)N_ATOM * WIDTH * 4);
  float*  AGG_F = (float*)carve((size_t)N_FRAG * WIDTH * 4);
  bf16_t* W1T   = (bf16_t*)carve((size_t)128 * 576 * 2);
  bf16_t* W2T   = (bf16_t*)carve((size_t)GPAD * 128 * 2);
  bf16_t* BIGWT = (bf16_t*)carve((size_t)320 * PDIM * 2);
  bf16_t* BIGPT = (bf16_t*)carve((size_t)320 * 320 * 2);
  bf16_t* LNWT  = (bf16_t*)carve((size_t)320 * 256 * 2);
  // node-update scratch aliases the P region (P is idle during node updates)
  size_t po = 0;
  auto carveP = [&](size_t bytes) -> char* {
    po = (po + 255) & ~(size_t)255;
    char* p = Pbase + po; po += bytes; return p;
  };
  bf16_t* AGG_BF = (bf16_t*)carveP((size_t)N_ATOM * WIDTH * 2);
  float*  PROJ   = (float*) carveP((size_t)N_ATOM * WIDTH * 4);
  bf16_t* COND   = (bf16_t*)carveP((size_t)N_ATOM * 256 * 2);
  float*  MOD    = (float*) carveP((size_t)N_ATOM * WIDTH * 4);

  // N must be a multiple of 64 (16x64 tile per wave)
  auto launch_gemm = [&](const bf16_t* A, int lda, const bf16_t* Bt, int ldb,
                         int M, int N, int K, const float* bias,
                         float* oF, bf16_t* oB, int ldo,
                         const int* dstIdx, float* agg, int Mvalid, int mode) {
    int Mt = M / 16, Ng = N / 64;
    int waves = Mt * Ng;
    int blocks = (waves + 7) / 8;
    gemm_bf16_wmma4<<<blocks, 256, 0, stream>>>(A, lda, Bt, ldb, Mt, Ng, K,
                                                bias, oF, oB, ldo, dstIdx, agg, Mvalid, mode);
  };

  struct Blk { const float* f; int st; const int* idx; const int* ind; };
  Blk ZB = { nullptr, 0, nullptr, nullptr };

  auto run_conv = [&](int E, int Epad, int Kpad,
                      const float* posP, const int* ip, const float* posM, const int* im,
                      const float* triRef,
                      Blk b0, Blk b1, Blk b2, Blk b3, int nb,
                      const float* hsrc, const int* srcIdx,
                      const int* dstIdx, float* agg, int pbase) {
    int edgeDim = (triRef ? 33 : 16) + nb * 128;
    transpose_pad<<<(128 * Kpad + 255) / 256, 256, 0, stream>>>(W1T, F(pbase + 0), edgeDim, 128, 128, Kpad);
    transpose_pad<<<(GPAD * 128 + 255) / 256, 256, 0, stream>>>(W2T, F(pbase + 2), 128, GDIM, GPAD, 128);
    build_bigwt<<<(320 * PDIM + 255) / 256, 256, 0, stream>>>(BIGWT,
        F(pbase + 3), F(pbase + 4), F(pbase + 5), F(pbase + 6), F(pbase + 7),
        F(pbase + 8), F(pbase + 9), F(pbase + 10), F(pbase + 11));
    build_es_kernel<<<Epad, 128, 0, stream>>>(ES, Kpad, E, posP, ip, posM, im, triRef, NVEC,
        b0.f, b0.st, b0.idx, b0.ind,  b1.f, b1.st, b1.idx, b1.ind,
        b2.f, b2.st, b2.idx, b2.ind,  b3.f, b3.st, b3.idx, b3.ind, nb);
    launch_gemm(ES, Kpad, W1T, Kpad, Epad, 128, Kpad, F(pbase + 1),
                nullptr, HID, 128, nullptr, nullptr, 0, MB_BIAS | MB_SILU);
    launch_gemm(HID, 128, W2T, 128, Epad, GPAD, 128, nullptr,
                nullptr, G, GPAD, nullptr, nullptr, 0, 0);
    build_P_kernel<<<Epad, 128, 0, stream>>>(P, E, hsrc, srcIdx, G, NVEC);
    launch_gemm(P, PDIM, BIGWT, PDIM, Epad, WIDTH, PDIM, nullptr,
                nullptr, nullptr, 0, dstIdx, agg, E, MB_SCAT);
  };

  // ================= phase 1: atom-destination convs =================
  hipMemsetAsync(AGG_A, 0, (size_t)N_ATOM * WIDTH * 4, stream);

  run_conv(40000, 40000, 416, apos, II(I_RADST), ppos, II(I_RASRC), nullptr,
           Blk{hprot,WIDTH,II(I_RASRC),nullptr}, Blk{hatom,WIDTH,II(I_RADST),nullptr},
           Blk{tea,128,II(I_RADST),nullptr}, ZB, 3,
           hprot, II(I_RASRC), II(I_RADST), AGG_A, I_RAP);

  run_conv(30000, 30000, 544, apos, II(I_BDST), apos, II(I_BSRC), nullptr,
           Blk{hatom,WIDTH,II(I_BSRC),nullptr}, Blk{hatom,WIDTH,II(I_BDST),nullptr},
           Blk{hfrag,WIDTH,II(I_BDST),fragid}, Blk{tea,128,II(I_BDST),nullptr}, 4,
           hatom, II(I_BSRC), II(I_BDST), AGG_A, I_AAP);

  run_conv(30000, 30000, 576, apos, II(I_TRIDST), apos, II(I_TRISRC), F(I_TRIREF),
           Blk{hatom,WIDTH,II(I_TRISRC),nullptr}, Blk{hatom,WIDTH,II(I_TRIDST),nullptr},
           Blk{hfrag,WIDTH,II(I_TRIDST),fragid}, Blk{tea,128,II(I_TRIDST),nullptr}, 4,
           hatom, II(I_TRISRC), II(I_TRIDST), AGG_A, I_TRIP);

  // ---- atom update: proj GEMM + mod GEMM + adaLN ----
  cast_f32_bf16<<<((N_ATOM * WIDTH) + 255) / 256, 256, 0, stream>>>(AGG_BF, AGG_A, N_ATOM * WIDTH);
  build_bigpt<<<(320 * 320 + 255) / 256, 256, 0, stream>>>(BIGPT, F(I_APROJ+0), F(I_APROJ+1), F(I_APROJ+2));
  launch_gemm(AGG_BF, WIDTH, BIGPT, WIDTH, N_ATOM, WIDTH, WIDTH, nullptr,
              PROJ, nullptr, WIDTH, nullptr, nullptr, 0, MB_F32);
  build_cond_atom<<<((N_ATOM * 256) + 255) / 256, 256, 0, stream>>>(COND, tea, hfrag, fragid, N_ATOM);
  transpose_pad<<<(320 * 256 + 255) / 256, 256, 0, stream>>>(LNWT, F(I_ALN), 256, 320, 320, 256);
  launch_gemm(COND, 256, LNWT, 256, N_ATOM, WIDTH, 256, F(I_ALN + 1),
              MOD, nullptr, WIDTH, nullptr, nullptr, 0, MB_BIAS | MB_F32);
  node_update_kernel<<<N_ATOM, 128, 0, stream>>>(hatom, PROJ, MOD, outAtom);

  // ================= phase 2: frag-destination convs =================
  hipMemsetAsync(AGG_F, 0, (size_t)N_FRAG * WIDTH * 4, stream);

  // af conv: sources are the UPDATED atoms (read back from d_out)
  run_conv(16000, 16000, 416, apos, nullptr, tfr, fragid, nullptr,
           Blk{outAtom,WIDTH,nullptr,nullptr}, Blk{hfrag,WIDTH,nullptr,fragid},
           Blk{tea,128,nullptr,nullptr}, ZB, 3,
           outAtom, nullptr, fragid, AGG_F, I_AFP);

  run_conv(15000, 15008, 416, tfr, II(I_PFDST), ppos, II(I_PFSRC), nullptr,
           Blk{hprot,WIDTH,II(I_PFSRC),nullptr}, Blk{hfrag,WIDTH,II(I_PFDST),nullptr},
           Blk{tef,128,II(I_PFDST),nullptr}, ZB, 3,
           hprot, II(I_PFSRC), II(I_PFDST), AGG_F, I_PFP);

  run_conv(8000, 8000, 416, tfr, II(I_FFDST), tfr, II(I_FFSRC), nullptr,
           Blk{hfrag,WIDTH,II(I_FFSRC),nullptr}, Blk{hfrag,WIDTH,II(I_FFDST),nullptr},
           Blk{tef,128,II(I_FFDST),nullptr}, ZB, 3,
           hfrag, II(I_FFSRC), II(I_FFDST), AGG_F, I_FFP);

  // ---- frag update ----
  cast_f32_bf16<<<((N_FRAG * WIDTH) + 255) / 256, 256, 0, stream>>>(AGG_BF, AGG_F, N_FRAG * WIDTH);
  build_bigpt<<<(320 * 320 + 255) / 256, 256, 0, stream>>>(BIGPT, F(I_FPROJ+0), F(I_FPROJ+1), F(I_FPROJ+2));
  launch_gemm(AGG_BF, WIDTH, BIGPT, WIDTH, N_FRAG, WIDTH, WIDTH, nullptr,
              PROJ, nullptr, WIDTH, nullptr, nullptr, 0, MB_F32);
  cast_f32_bf16<<<((N_FRAG * 128) + 255) / 256, 256, 0, stream>>>(COND, tef, N_FRAG * 128);
  transpose_pad<<<(320 * 128 + 255) / 256, 256, 0, stream>>>(LNWT, F(I_FLN), 128, 320, 320, 128);
  launch_gemm(COND, 128, LNWT, 128, N_FRAG, WIDTH, 128, F(I_FLN + 1),
              MOD, nullptr, WIDTH, nullptr, nullptr, 0, MB_BIAS | MB_F32);
  node_update_kernel<<<N_FRAG, 128, 0, stream>>>(hfrag, PROJ, MOD, outFrag);
}
